// XSimGCL_77567109366285
// MI455X (gfx1250) — matrix operand order repaired and checked by reading, not compiled, not run on hardware
//
#include <hip/hip_runtime.h>
#include <hip/hip_bf16.h>

#define N_USERS 50000
#define N_ITEMS 50000
#define N_TOT   100000
#define D_EMB   64
#define BATCH   4096
#define EPS_P   0.2f
#define INV_T   5.0f      // 1/TEMP, TEMP=0.2
#define CL_RATE 0.2f
#define DECAY   0.0001f

typedef _Float16 f16;
typedef __attribute__((ext_vector_type(16))) _Float16 v16h;
typedef __attribute__((ext_vector_type(8)))  _Float16 v8h;
typedef __attribute__((ext_vector_type(8)))  float    v8f;

union V16U { v16h v; v8h h[2]; };

// ---------------- elementwise helpers ----------------

__global__ void k_zero_f4(float4* __restrict__ p, int n4) {
    int i = blockIdx.x * blockDim.x + threadIdx.x;
    if (i < n4) p[i] = make_float4(0.f, 0.f, 0.f, 0.f);
}

__global__ void k_zero_i(int* __restrict__ p, int n) {
    int i = blockIdx.x * blockDim.x + threadIdx.x;
    if (i < n) p[i] = 0;
}

__global__ void k_concat(const float4* __restrict__ ue, const float4* __restrict__ ie,
                         float4* __restrict__ out) {
    int i = blockIdx.x * blockDim.x + threadIdx.x;
    const int nu4 = (N_USERS * D_EMB) / 4;
    const int nt4 = (N_TOT * D_EMB) / 4;
    if (i >= nt4) return;
    out[i] = (i < nu4) ? ue[i] : ie[i - nu4];
}

// mean of 3 layer outputs, written into d (reads d first)
__global__ void k_mean3(const float4* __restrict__ a, const float4* __restrict__ b,
                        float4* __restrict__ d) {
    int i = blockIdx.x * blockDim.x + threadIdx.x;
    const int nt4 = (N_TOT * D_EMB) / 4;
    if (i >= nt4) return;
    float4 va = a[i], vb = b[i], vd = d[i];
    const float s = 1.0f / 3.0f;
    d[i] = make_float4((va.x + vb.x + vd.x) * s, (va.y + vb.y + vd.y) * s,
                       (va.z + vb.z + vd.z) * s, (va.w + vb.w + vd.w) * s);
}

// ---------------- SpMM: y[rows[e]] += vals[e] * x[cols[e]] ----------------
// 16 threads per edge, each covers 4 contiguous dims (float4 gather, 4 atomics)
__global__ void k_spmm(const int* __restrict__ rows, const int* __restrict__ cols,
                       const float* __restrict__ vals, const float* __restrict__ x,
                       float* __restrict__ y, int E) {
    long long gid = (long long)blockIdx.x * blockDim.x + threadIdx.x;
    long long e = gid >> 4;
    if (e >= E) return;
    int part = (int)(gid & 15);
    int r = rows[e], c = cols[e];
    float v = vals[e];
    float4 xc = ((const float4*)(x + (size_t)c * D_EMB))[part];
    float* yp = y + (size_t)r * D_EMB + part * 4;
    atomicAdd(yp + 0, v * xc.x);
    atomicAdd(yp + 1, v * xc.y);
    atomicAdd(yp + 2, v * xc.z);
    atomicAdd(yp + 3, v * xc.w);
}

// ---------------- perturb: x += sign(x) * l2norm(noise_row) * EPS ----------------
// one wave per row, 2 dims per lane
__global__ void k_perturb(float* __restrict__ buf, const float* __restrict__ noise) {
    int row = blockIdx.x * (blockDim.x >> 5) + (threadIdx.x >> 5);
    int lane = threadIdx.x & 31;
    if (row >= N_TOT) return;
    size_t off = (size_t)row * D_EMB;
    float n0 = noise[off + lane], n1 = noise[off + lane + 32];
    float s = n0 * n0 + n1 * n1;
    for (int m = 1; m < 32; m <<= 1) s += __shfl_xor(s, m, 32);
    float inv = 1.0f / fmaxf(sqrtf(s), 1e-12f);
    float x0 = buf[off + lane], x1 = buf[off + lane + 32];
    float g0 = (x0 > 0.f) ? 1.f : ((x0 < 0.f) ? -1.f : 0.f);
    float g1 = (x1 > 0.f) ? 1.f : ((x1 < 0.f) ? -1.f : 0.f);
    buf[off + lane]      = x0 + g0 * (n0 * inv) * EPS_P;
    buf[off + lane + 32] = x1 + g1 * (n1 * inv) * EPS_P;
}

// ---------------- unique via flag array ----------------
__global__ void k_unique(const int* __restrict__ ids, int n, int* __restrict__ flags,
                         int* __restrict__ list, int* __restrict__ cnt) {
    int b = blockIdx.x * blockDim.x + threadIdx.x;
    if (b >= n) return;
    int u = ids[b];
    if (atomicExch(&flags[u], 1) == 0) {
        int p = atomicAdd(cnt, 1);
        list[p] = u;
    }
}

// ---------------- normalize, convert to f16, positive score ----------------
// one wave per unique element k; 2 dims per lane.
// Rows U..ceil16(U)-1 are zero-filled so the WMMA kernel can load fragments
// unconditionally (padded columns are masked out of the exp-sum consumer-side).
__global__ void k_prep_cl(const int* __restrict__ list, const int* __restrict__ cnt,
                          const float* __restrict__ light, const float* __restrict__ ecl,
                          int rowOffset, f16* __restrict__ V1, f16* __restrict__ V2,
                          float* __restrict__ posArr, float* __restrict__ ttlArr) {
    int k = blockIdx.x * (blockDim.x >> 5) + (threadIdx.x >> 5);
    int lane = threadIdx.x & 31;
    int U = *cnt;
    int padded = (U + 15) & ~15;
    if (padded > BATCH) padded = BATCH;
    if (k >= padded) return;
    size_t ko = (size_t)k * D_EMB;
    if (k >= U) {                 // zero padding rows
        V1[ko + lane] = (f16)0.f; V1[ko + lane + 32] = (f16)0.f;
        V2[ko + lane] = (f16)0.f; V2[ko + lane + 32] = (f16)0.f;
        return;
    }
    size_t row = (size_t)(rowOffset + list[k]) * D_EMB;
    float a0 = light[row + lane], a1 = light[row + lane + 32];
    float b0 = ecl[row + lane],   b1 = ecl[row + lane + 32];
    float sa = a0 * a0 + a1 * a1, sb = b0 * b0 + b1 * b1;
    for (int m = 1; m < 32; m <<= 1) {
        sa += __shfl_xor(sa, m, 32);
        sb += __shfl_xor(sb, m, 32);
    }
    float na = fmaxf(sqrtf(sa), 1e-12f), nb = fmaxf(sqrtf(sb), 1e-12f);
    a0 /= na; a1 /= na; b0 /= nb; b1 /= nb;
    float d = a0 * b0 + a1 * b1;
    for (int m = 1; m < 32; m <<= 1) d += __shfl_xor(d, m, 32);
    V1[ko + lane] = (f16)a0; V1[ko + lane + 32] = (f16)a1;
    V2[ko + lane] = (f16)b0; V2[ko + lane + 32] = (f16)b1;
    if (lane == 0) {
        posArr[k] = __expf(d * INV_T);
        ttlArr[k] = 0.0f;
    }
}

// ---------------- WMMA: ttl[i] += sum_j exp(dot(v1_i, v2_j)/T) ----------------
// One wave per 16x64 output strip: one A fragment (K=64, two chunks) reused
// across 4 B tiles -> 8 v_wmma_f32_16x16x32_f16 per wave. V1/V2 are zero-padded
// to a multiple of 16 rows, so fragment loads are unconditional; padded columns
// (j >= U) are masked out of the exp accumulation. Row sums are reduced across
// the 16 lanes sharing a row before a single atomic per row per strip.
__global__ void k_infonce_ttl(const f16* __restrict__ V1, const f16* __restrict__ V2,
                              const int* __restrict__ cnt, float* __restrict__ ttl) {
    int U = *cnt;
    int tiles = (U + 15) >> 4;
    int wave = threadIdx.x >> 5;
    int lane = threadIdx.x & 31;
    int strip = blockIdx.x * 4 + wave;       // 4 waves per block
    int ti = blockIdx.y;
    int tjbase = strip * 4;
    if (ti >= tiles || tjbase >= tiles) return;   // wave-uniform exit

    int m  = lane & 15;
    int hi = (lane >> 4) & 1;

    // A fragments for K chunks 0 and 1 (this code's contiguous k order, same as B:
    // lanes 0-15 hold v8h groups {c*4+0, c*4+1}; lanes 16-31 hold {c*4+2, c*4+3})
    int arow = ti * 16 + m;
    const v8h* pa = (const v8h*)(V1 + (size_t)arow * D_EMB);
    V16U a0, a1;
    a0.h[0] = pa[2 * hi];     a0.h[1] = pa[2 * hi + 1];
    a1.h[0] = pa[4 + 2 * hi]; a1.h[1] = pa[4 + 2 * hi + 1];

    v8f esum = {};
    #pragma unroll
    for (int t = 0; t < 4; ++t) {
        int tj = tjbase + t;
        if (tj < tiles) {                    // wave-uniform guard
            // B fragment (this code's contiguous k order, same as A: lanes 0-15 K=c*32+0..15,
            // lanes 16-31 K=c*32+16..31)
            int brow = tj * 16 + m;
            const v8h* pb = (const v8h*)(V2 + (size_t)brow * D_EMB);
            int g0 = hi << 1;
            V16U b0, b1;
            b0.h[0] = pb[g0];     b0.h[1] = pb[g0 + 1];
            b1.h[0] = pb[4 + g0]; b1.h[1] = pb[4 + g0 + 1];
            v8f c = {};
            c = __builtin_amdgcn_wmma_f32_16x16x32_f16(false, a0.v, false, b0.v,
                                                       (short)0, c, false, false);
            c = __builtin_amdgcn_wmma_f32_16x16x32_f16(false, a1.v, false, b1.v,
                                                       (short)0, c, false, false);
            int j = tj * 16 + m;
            float mask = (j < U) ? 1.0f : 0.0f;
            #pragma unroll
            for (int r = 0; r < 8; ++r)
                esum[r] += mask * __expf(c[r] * INV_T);
        }
    }

    // C/D layout: VGPR r, lane L -> M = r + (L>=16 ? 8 : 0), N = L%16
    int ibase = ti * 16 + (hi << 3);
    #pragma unroll
    for (int r = 0; r < 8; ++r) {
        float e = esum[r];
        for (int mm = 1; mm < 16; mm <<= 1) e += __shfl_xor(e, mm, 32);
        if (m == 0) atomicAdd(&ttl[ibase + r], e);   // rows >= U land in scratch, never read
    }
}

// ---------------- InfoNCE mean: acc += mean_k(-log(pos/ttl + 1e-5)) ----------------
__global__ void k_cl_reduce(const float* __restrict__ posArr, const float* __restrict__ ttlArr,
                            const int* __restrict__ cnt, float* __restrict__ acc) {
    __shared__ float sm[256];
    int U = *cnt;
    float s = 0.f;
    for (int k = threadIdx.x; k < U; k += blockDim.x)
        s += -__logf(posArr[k] / ttlArr[k] + 1e-5f);
    sm[threadIdx.x] = s;
    __syncthreads();
    for (int st = 128; st > 0; st >>= 1) {
        if ((int)threadIdx.x < st) sm[threadIdx.x] += sm[threadIdx.x + st];
        __syncthreads();
    }
    if (threadIdx.x == 0 && U > 0) atomicAdd(acc, sm[0] / (float)U);
}

// ---------------- BPR + regularizer: one wave per batch element ----------------
__global__ void k_bpr(const int* __restrict__ users, const int* __restrict__ pos,
                      const int* __restrict__ neg, const float* __restrict__ light,
                      const float* __restrict__ ue0, const float* __restrict__ ie0,
                      float* __restrict__ accMaxi, float* __restrict__ accReg) {
    int b = blockIdx.x * (blockDim.x >> 5) + (threadIdx.x >> 5);
    int lane = threadIdx.x & 31;
    if (b >= BATCH) return;
    int u = users[b], p = pos[b], n = neg[b];
    const float* Ur = light + (size_t)u * D_EMB;
    const float* Pr = light + (size_t)(N_USERS + p) * D_EMB;
    const float* Nr = light + (size_t)(N_USERS + n) * D_EMB;
    float u0 = Ur[lane], u1 = Ur[lane + 32];
    float p0 = Pr[lane], p1 = Pr[lane + 32];
    float n0 = Nr[lane], n1 = Nr[lane + 32];
    float su = u0 * u0 + u1 * u1, sp = p0 * p0 + p1 * p1, sn = n0 * n0 + n1 * n1;
    float dp = u0 * p0 + u1 * p1, dn = u0 * n0 + u1 * n1;
    const float* Ue = ue0 + (size_t)u * D_EMB;
    const float* Pe = ie0 + (size_t)p * D_EMB;
    const float* Ne = ie0 + (size_t)n * D_EMB;
    float rg = Ue[lane] * Ue[lane] + Ue[lane + 32] * Ue[lane + 32]
             + Pe[lane] * Pe[lane] + Pe[lane + 32] * Pe[lane + 32]
             + Ne[lane] * Ne[lane] + Ne[lane + 32] * Ne[lane + 32];
    for (int mm = 1; mm < 32; mm <<= 1) {
        su += __shfl_xor(su, mm, 32); sp += __shfl_xor(sp, mm, 32);
        sn += __shfl_xor(sn, mm, 32); dp += __shfl_xor(dp, mm, 32);
        dn += __shfl_xor(dn, mm, 32); rg += __shfl_xor(rg, mm, 32);
    }
    if (lane == 0) {
        float nu = fmaxf(sqrtf(su), 1e-12f);
        float np = fmaxf(sqrtf(sp), 1e-12f);
        float nn = fmaxf(sqrtf(sn), 1e-12f);
        float ps = dp / (nu * np), ns = dn / (nu * nn);
        float maxi = __logf(1.0f / (1.0f + __expf(-(ps - ns))) + 1e-6f);
        atomicAdd(accMaxi, maxi);
        atomicAdd(accReg, rg);
    }
}

__global__ void k_final(const float* __restrict__ acc, float* __restrict__ out) {
    if (threadIdx.x == 0) {
        out[0] = -acc[0] / (float)BATCH;                       // mf_loss
        out[1] = CL_RATE * acc[2];                             // cl_loss
        out[2] = DECAY * (0.5f * acc[1]) / (float)BATCH;       // reg_loss
    }
}

// ---------------- launch ----------------
extern "C" void kernel_launch(void* const* d_in, const int* in_sizes, int n_in,
                              void* d_out, int out_size, void* d_ws, size_t ws_size,
                              hipStream_t stream) {
    const int*   d_users = (const int*)d_in[0];
    const int*   d_pos   = (const int*)d_in[1];
    const int*   d_neg   = (const int*)d_in[2];
    const float* d_ue    = (const float*)d_in[3];
    const float* d_ie    = (const float*)d_in[4];
    const int*   d_rows  = (const int*)d_in[5];
    const int*   d_cols  = (const int*)d_in[6];
    const float* d_vals  = (const float*)d_in[7];
    const float* d_noise = (const float*)d_in[8];
    const int E = in_sizes[5];

    // workspace layout
    const size_t EMB = (size_t)N_TOT * D_EMB * sizeof(float);   // 25.6 MB
    char* p = (char*)d_ws;
    float* bufD = (float*)p; p += EMB;           // x0, then layer-3 out, then light_out
    float* bufA = (float*)p; p += EMB;           // layer-1 out == emb_cl
    float* bufB = (float*)p; p += EMB;           // layer-2 out
    f16* V1 = (f16*)p;       p += (size_t)BATCH * D_EMB * sizeof(f16);
    f16* V2 = (f16*)p;       p += (size_t)BATCH * D_EMB * sizeof(f16);
    float* posArr = (float*)p; p += BATCH * sizeof(float);
    float* ttlArr = (float*)p; p += BATCH * sizeof(float);
    int* flags = (int*)p;    p += N_USERS * sizeof(int);
    int* list  = (int*)p;    p += BATCH * sizeof(int);
    int* cnt   = (int*)p;    p += 64;
    float* acc = (float*)p;  p += 64;            // [0]=sum maxi, [1]=sum reg, [2]=sum infonce

    const int nt4 = (N_TOT * D_EMB) / 4;
    const int gElem = (nt4 + 255) / 256;
    const int gSpmm = (int)(((long long)E * 16 + 255) / 256);
    const int gRow  = (N_TOT + 7) / 8;           // 8 waves/block, 1 row/wave

    k_zero_i<<<1, 32, 0, stream>>>((int*)acc, 4);

    // x0 = concat(user_emb, item_emb)
    k_concat<<<gElem, 256, 0, stream>>>((const float4*)d_ue, (const float4*)d_ie,
                                        (float4*)bufD);

    // layer 1: A = spmm(D) + perturb(noise[0])
    k_zero_f4<<<gElem, 256, 0, stream>>>((float4*)bufA, nt4);
    k_spmm<<<gSpmm, 256, 0, stream>>>(d_rows, d_cols, d_vals, bufD, bufA, E);
    k_perturb<<<gRow, 256, 0, stream>>>(bufA, d_noise);

    // layer 2: B = spmm(A) + perturb(noise[1])
    k_zero_f4<<<gElem, 256, 0, stream>>>((float4*)bufB, nt4);
    k_spmm<<<gSpmm, 256, 0, stream>>>(d_rows, d_cols, d_vals, bufA, bufB, E);
    k_perturb<<<gRow, 256, 0, stream>>>(bufB, d_noise + (size_t)1 * N_TOT * D_EMB);

    // layer 3: D = spmm(B) + perturb(noise[2])   (x0 dead, reuse)
    k_zero_f4<<<gElem, 256, 0, stream>>>((float4*)bufD, nt4);
    k_spmm<<<gSpmm, 256, 0, stream>>>(d_rows, d_cols, d_vals, bufB, bufD, E);
    k_perturb<<<gRow, 256, 0, stream>>>(bufD, d_noise + (size_t)2 * N_TOT * D_EMB);

    // light_out = (A + B + D) / 3 -> D
    k_mean3<<<gElem, 256, 0, stream>>>((const float4*)bufA, (const float4*)bufB,
                                       (float4*)bufD);

    dim3 gW(16, 256);        // 4 waves/block, each wave = 16x64 strip
    const int gWave = (BATCH + 7) / 8;

    // ---- InfoNCE over unique users ----
    k_zero_i<<<(N_USERS + 255) / 256, 256, 0, stream>>>(flags, N_USERS);
    k_zero_i<<<1, 32, 0, stream>>>(cnt, 1);
    k_unique<<<(BATCH + 255) / 256, 256, 0, stream>>>(d_users, BATCH, flags, list, cnt);
    k_prep_cl<<<gWave, 256, 0, stream>>>(list, cnt, bufD, bufA, 0, V1, V2, posArr, ttlArr);
    k_infonce_ttl<<<gW, 128, 0, stream>>>(V1, V2, cnt, ttlArr);
    k_cl_reduce<<<1, 256, 0, stream>>>(posArr, ttlArr, cnt, &acc[2]);

    // ---- InfoNCE over unique pos_items ----
    k_zero_i<<<(N_ITEMS + 255) / 256, 256, 0, stream>>>(flags, N_ITEMS);
    k_zero_i<<<1, 32, 0, stream>>>(cnt, 1);
    k_unique<<<(BATCH + 255) / 256, 256, 0, stream>>>(d_pos, BATCH, flags, list, cnt);
    k_prep_cl<<<gWave, 256, 0, stream>>>(list, cnt, bufD, bufA, N_USERS, V1, V2, posArr, ttlArr);
    k_infonce_ttl<<<gW, 128, 0, stream>>>(V1, V2, cnt, ttlArr);
    k_cl_reduce<<<1, 256, 0, stream>>>(posArr, ttlArr, cnt, &acc[2]);

    // ---- BPR + reg ----
    k_bpr<<<gWave, 256, 0, stream>>>(d_users, d_pos, d_neg, bufD, d_ue, d_ie,
                                     &acc[0], &acc[1]);

    k_final<<<1, 32, 0, stream>>>(acc, (float*)d_out);
}
